// Model_81870666596985
// MI455X (gfx1250) — matrix operand order repaired and checked by reading, not compiled, not run on hardware
//
#include <hip/hip_runtime.h>
#include <hip/hip_bf16.h>
#include <stdint.h>

// ---------------------------------------------------------------------------
// MoE (4 experts, top-3 gating) forward for MI455X / gfx1250.
// Heavy math done with v_wmma_f32_16x16x32_bf16 (wave32 WMMA).
// M-tile = 32: each wave computes two 16x16 output subtiles per B-fragment,
// halving L2 weight traffic vs. the M=16 version (32 FLOP/byte from L2).
// ---------------------------------------------------------------------------

typedef __attribute__((ext_vector_type(16))) __bf16          v16bf;
typedef __attribute__((ext_vector_type(8)))  float           v8f;
typedef __attribute__((ext_vector_type(8)))  unsigned short  ushort8;

#define SEQ_LEN   64
#define REPR_DIM  512
#define DTOT      (SEQ_LEN * REPR_DIM)   // 32768
#define BATCH     1024
#define NEXP      4
#define NROWS     (BATCH * SEQ_LEN)      // 65536 rows of r
#define MTILE     32                     // rows per workgroup (divides SEQ_LEN)
#define LOSS_COEF 0.01f

__device__ __forceinline__ unsigned short f32_to_bf16_rne(float f) {
    union { float f; uint32_t u; } c; c.f = f;
    uint32_t u = c.u;
    u += 0x7FFFu + ((u >> 16) & 1u);          // round-to-nearest-even
    return (unsigned short)(u >> 16);
}

__device__ __forceinline__ float gelu_exact(float v) {
    return 0.5f * v * (1.0f + erff(v * 0.70710678118654752f));
}

union Frag16 { ushort8 h[2]; v16bf v; };

__device__ __forceinline__ v8f wmma_bf16(v16bf a, v16bf b, v8f c) {
    // (neg_a, A, neg_b, B, c_mod, C, reuse_a, reuse_b)
    return __builtin_amdgcn_wmma_f32_16x16x32_bf16(
        false, a, false, b, (short)0, c, false, false);
}

// ---------------------------------------------------------------------------
// Stage 1: convert fp32 weight [K][N] -> bf16 transposed [N][K] so a B-matrix
// fragment (fixed column n, contiguous k) is two contiguous 16B loads.
// ---------------------------------------------------------------------------
__global__ void wt_convert(const float* __restrict__ src,
                           unsigned short* __restrict__ dst,
                           int K, int N) {
    int idx = blockIdx.x * blockDim.x + threadIdx.x;
    if (idx >= K * N) return;
    int n = idx % N;
    int k = idx / N;
    dst[(size_t)n * K + k] = f32_to_bf16_rne(src[(size_t)k * N + n]);
}

// ---------------------------------------------------------------------------
// Stage 2: gating. One 256-thread block per batch row: 4 dot products of
// length 32768, then top-3-of-4 softmax gates (excluded expert = argmin).
// ---------------------------------------------------------------------------
__global__ void gating_kernel(const float* __restrict__ x,
                              const float* __restrict__ wg,   // [D][4]
                              float* __restrict__ gates) {    // [B][4]
    __shared__ float4 red[256];
    const int b   = blockIdx.x;
    const int tid = threadIdx.x;
    const float*  xr  = x + (size_t)b * DTOT;
    const float4* wg4 = (const float4*)wg;
    float4 acc = make_float4(0.f, 0.f, 0.f, 0.f);
    for (int d = tid; d < DTOT; d += 256) {
        const float  xv = xr[d];
        const float4 w  = wg4[d];
        acc.x += xv * w.x; acc.y += xv * w.y;
        acc.z += xv * w.z; acc.w += xv * w.w;
    }
    red[tid] = acc;
    __syncthreads();
    for (int s = 128; s > 0; s >>= 1) {
        if (tid < s) {
            float4 o = red[tid + s]; float4 m = red[tid];
            m.x += o.x; m.y += o.y; m.z += o.z; m.w += o.w;
            red[tid] = m;
        }
        __syncthreads();
    }
    if (tid == 0) {
        float l[4] = {red[0].x, red[0].y, red[0].z, red[0].w};
        int mini = 0;
        for (int e = 1; e < 4; ++e) if (l[e] <= l[mini]) mini = e;  // matches top_k tiebreak
        float maxv = -3.4e38f;
        for (int e = 0; e < 4; ++e) if (e != mini && l[e] > maxv) maxv = l[e];
        float ex[4]; float s = 0.f;
        for (int e = 0; e < 4; ++e) {
            ex[e] = (e == mini) ? 0.f : expf(l[e] - maxv);
            s += ex[e];
        }
        for (int e = 0; e < 4; ++e) gates[b * 4 + e] = ex[e] / s;
    }
}

// ---------------------------------------------------------------------------
// Stage 3: load-balance loss from the gate matrix (no atomics needed).
// ---------------------------------------------------------------------------
__global__ void loss_kernel(const float* __restrict__ gates,
                            float* __restrict__ loss_out) {
    __shared__ float4 rimp[256];
    __shared__ float4 rld[256];
    const int tid = threadIdx.x;
    const float4* g4 = (const float4*)gates;
    float4 imp = make_float4(0,0,0,0), ld = make_float4(0,0,0,0);
    for (int b = tid; b < BATCH; b += 256) {
        float4 g = g4[b];
        imp.x += g.x; imp.y += g.y; imp.z += g.z; imp.w += g.w;
        ld.x += (g.x > 0.f) ? 1.f : 0.f;
        ld.y += (g.y > 0.f) ? 1.f : 0.f;
        ld.z += (g.z > 0.f) ? 1.f : 0.f;
        ld.w += (g.w > 0.f) ? 1.f : 0.f;
    }
    rimp[tid] = imp; rld[tid] = ld;
    __syncthreads();
    for (int s = 128; s > 0; s >>= 1) {
        if (tid < s) {
            float4 a = rimp[tid], b2 = rimp[tid + s];
            a.x += b2.x; a.y += b2.y; a.z += b2.z; a.w += b2.w; rimp[tid] = a;
            float4 c = rld[tid],  d  = rld[tid + s];
            c.x += d.x; c.y += d.y; c.z += d.z; c.w += d.w;   rld[tid]  = c;
        }
        __syncthreads();
    }
    if (tid == 0) {
        float4 vi = rimp[0], vl = rld[0];
        float mi = (vi.x + vi.y + vi.z + vi.w) * 0.25f;
        float vari = ((vi.x-mi)*(vi.x-mi) + (vi.y-mi)*(vi.y-mi) +
                      (vi.z-mi)*(vi.z-mi) + (vi.w-mi)*(vi.w-mi)) / 3.0f;
        float ml = (vl.x + vl.y + vl.z + vl.w) * 0.25f;
        float varl = ((vl.x-ml)*(vl.x-ml) + (vl.y-ml)*(vl.y-ml) +
                      (vl.z-ml)*(vl.z-ml) + (vl.w-ml)*(vl.w-ml)) / 3.0f;
        loss_out[0] = (vari / (mi * mi + 1e-10f) + varl / (ml * ml + 1e-10f)) * LOSS_COEF;
    }
}

// ---------------------------------------------------------------------------
// Stage 4: fused expert pipeline. One block (8 wave32) per 32-row tile of r.
// Each wave owns one 16-col N-tile and computes BOTH 16-row M-subtiles with
// a single B-fragment load (2 WMMAs per B fetch). Intermediates in LDS; y
// accumulated in f32 LDS with the residual folded in.  LDS = 224 KB.
// ---------------------------------------------------------------------------
struct ExpertW {
    const unsigned short* w1[NEXP];  // [bn][512]  bf16
    const unsigned short* w2[NEXP];  // [bn][bn]   bf16
    const unsigned short* w3[NEXP];  // [512][bn]  bf16
    const float* b1[NEXP];
    const float* b2[NEXP];
    const float* b3[NEXP];
};

__launch_bounds__(256, 1)
__global__ void moe_expert_kernel(const float* __restrict__ x,
                                  const float* __restrict__ gates,
                                  float* __restrict__ y,
                                  ExpertW ew) {
    __shared__ unsigned short A0[MTILE * REPR_DIM];   // 32 KB  bf16 x-tile [32][512]
    __shared__ unsigned short H1[MTILE * 1024];       // 64 KB  bf16 hidden1 [32][bn]
    __shared__ unsigned short H2[MTILE * 1024];       // 64 KB  bf16 hidden2 [32][bn]
    __shared__ float          YACC[MTILE * REPR_DIM]; // 64 KB  f32 y accumulator

    const int tid  = threadIdx.x;
    const int lane = tid & 31;
    const int wave = tid >> 5;    // 0..7
    const int half = lane >> 4;   // which 16-lane half of the wave
    const int l15  = lane & 15;

    const int m0    = blockIdx.x * MTILE;  // 32-row tiles never straddle a token
    const int token = m0 / SEQ_LEN;

    const float4 gv = *(const float4*)(gates + (size_t)token * 4);
    const float gate[NEXP] = {gv.x, gv.y, gv.z, gv.w};

    // ---- stage x tile: f32 residual into YACC, bf16 A operand into A0 ----
    {
        const float4* xin  = (const float4*)(x + (size_t)m0 * REPR_DIM);
        float4*       yac4 = (float4*)YACC;
        uint32_t*     a32  = (uint32_t*)A0;
        for (int i = tid; i < MTILE * REPR_DIM / 4; i += 256) {
            float4 v = xin[i];
            yac4[i] = v;
            a32[2*i]   = (uint32_t)f32_to_bf16_rne(v.x) | ((uint32_t)f32_to_bf16_rne(v.y) << 16);
            a32[2*i+1] = (uint32_t)f32_to_bf16_rne(v.z) | ((uint32_t)f32_to_bf16_rne(v.w) << 16);
        }
    }
    __syncthreads();

    const int BN[NEXP] = {128, 256, 512, 1024};

    for (int e = 0; e < NEXP; ++e) {
        const int   bn = BN[e];
        const float g  = gate[e];

        // ---- Layer 1: H1 = gelu(A0[32x512] * W1[512xbn] + b1) ----
        for (int nt = wave; nt < (bn >> 4); nt += 8) {       // uniform per wave
            const int n = (nt << 4) + l15;
            v8f acc0 = {}, acc1 = {};
            const unsigned short* wp  = ew.w1[e] + (size_t)n * REPR_DIM + half * 8;
            const unsigned short* ap0 = &A0[l15 * REPR_DIM + half * 8];
            const unsigned short* ap1 = ap0 + 16 * REPR_DIM;
            for (int k0 = 0; k0 < REPR_DIM; k0 += 32) {
                Frag16 a0, a1, b;
                b.h[0]  = *(const ushort8*)(wp + k0);        // K: k0+h*8 .. +7
                b.h[1]  = *(const ushort8*)(wp + k0 + 16);
                a0.h[0] = *(const ushort8*)(ap0 + k0);       // K: k0+h*8 .. +7
                a0.h[1] = *(const ushort8*)(ap0 + k0 + 16);  // K: k0+16+h*8 .. +7
                a1.h[0] = *(const ushort8*)(ap1 + k0);
                a1.h[1] = *(const ushort8*)(ap1 + k0 + 16);
                __builtin_prefetch(wp + k0 + 256, 0, 1);
                acc0 = wmma_bf16(a0.v, b.v, acc0);
                acc1 = wmma_bf16(a1.v, b.v, acc1);
            }
            const float bias = ew.b1[e][n];
            #pragma unroll
            for (int j = 0; j < 8; ++j) {                    // C elem: M=j+8*half, N=l15
                H1[(j + half * 8)      * bn + n] = f32_to_bf16_rne(gelu_exact(acc0[j] + bias));
                H1[(j + half * 8 + 16) * bn + n] = f32_to_bf16_rne(gelu_exact(acc1[j] + bias));
            }
        }
        __syncthreads();

        // ---- Layer 2: H2 = gelu(H1[32xbn] * W2[bnxbn] + b2) ----
        for (int nt = wave; nt < (bn >> 4); nt += 8) {
            const int n = (nt << 4) + l15;
            v8f acc0 = {}, acc1 = {};
            const unsigned short* wp  = ew.w2[e] + (size_t)n * bn + half * 8;
            const unsigned short* ap0 = &H1[l15 * bn + half * 8];
            const unsigned short* ap1 = ap0 + 16 * bn;
            for (int k0 = 0; k0 < bn; k0 += 32) {
                Frag16 a0, a1, b;
                b.h[0]  = *(const ushort8*)(wp + k0);
                b.h[1]  = *(const ushort8*)(wp + k0 + 16);
                a0.h[0] = *(const ushort8*)(ap0 + k0);
                a0.h[1] = *(const ushort8*)(ap0 + k0 + 16);
                a1.h[0] = *(const ushort8*)(ap1 + k0);
                a1.h[1] = *(const ushort8*)(ap1 + k0 + 16);
                acc0 = wmma_bf16(a0.v, b.v, acc0);
                acc1 = wmma_bf16(a1.v, b.v, acc1);
            }
            const float bias = ew.b2[e][n];
            #pragma unroll
            for (int j = 0; j < 8; ++j) {
                H2[(j + half * 8)      * bn + n] = f32_to_bf16_rne(gelu_exact(acc0[j] + bias));
                H2[(j + half * 8 + 16) * bn + n] = f32_to_bf16_rne(gelu_exact(acc1[j] + bias));
            }
        }
        __syncthreads();

        // ---- Layer 3: YACC += g * (H2[32xbn] * W3[bnx512] + b3) ----
        for (int nt = wave; nt < (REPR_DIM >> 4); nt += 8) {
            const int n = (nt << 4) + l15;
            v8f acc0 = {}, acc1 = {};
            const unsigned short* wp  = ew.w3[e] + (size_t)n * bn + half * 8;
            const unsigned short* ap0 = &H2[l15 * bn + half * 8];
            const unsigned short* ap1 = ap0 + 16 * bn;
            for (int k0 = 0; k0 < bn; k0 += 32) {
                Frag16 a0, a1, b;
                b.h[0]  = *(const ushort8*)(wp + k0);
                b.h[1]  = *(const ushort8*)(wp + k0 + 16);
                a0.h[0] = *(const ushort8*)(ap0 + k0);
                a0.h[1] = *(const ushort8*)(ap0 + k0 + 16);
                a1.h[0] = *(const ushort8*)(ap1 + k0);
                a1.h[1] = *(const ushort8*)(ap1 + k0 + 16);
                acc0 = wmma_bf16(a0.v, b.v, acc0);
                acc1 = wmma_bf16(a1.v, b.v, acc1);
            }
            const float bias = ew.b3[e][n];
            #pragma unroll
            for (int j = 0; j < 8; ++j) {
                const int r0 = (j + half * 8) * REPR_DIM + n;        // unique per thread
                YACC[r0]                       += g * (acc0[j] + bias);
                YACC[r0 + 16 * REPR_DIM]       += g * (acc1[j] + bias);
            }
        }
        __syncthreads();
    }

    // ---- write y = x + sum_e g_e * h_e ----
    {
        const float4* yac4 = (const float4*)YACC;
        float4* yo = (float4*)(y + (size_t)m0 * REPR_DIM);
        for (int i = tid; i < MTILE * REPR_DIM / 4; i += 256) yo[i] = yac4[i];
    }
}

// ---------------------------------------------------------------------------
// Host launcher
// ---------------------------------------------------------------------------
extern "C" void kernel_launch(void* const* d_in, const int* in_sizes, int n_in,
                              void* d_out, int out_size, void* d_ws, size_t ws_size,
                              hipStream_t stream) {
    const float* x  = (const float*)d_in[0];
    const float* wg = (const float*)d_in[1];
    float* out = (float*)d_out;

    // Workspace layout: [gates 16KB][pad][bf16 weights ~6.7MB]
    char* ws = (char*)d_ws;
    float* gates = (float*)ws;
    unsigned short* wbuf = (unsigned short*)(ws + 32 * 1024);

    static const int BN[NEXP] = {128, 256, 512, 1024};
    ExpertW ew;
    size_t off = 0;
    for (int e = 0; e < NEXP; ++e) {
        const float* W1 = (const float*)d_in[2 + 6 * e + 0];
        const float* W2 = (const float*)d_in[2 + 6 * e + 2];
        const float* W3 = (const float*)d_in[2 + 6 * e + 4];
        ew.b1[e] = (const float*)d_in[2 + 6 * e + 1];
        ew.b2[e] = (const float*)d_in[2 + 6 * e + 3];
        ew.b3[e] = (const float*)d_in[2 + 6 * e + 5];
        const int bn = BN[e];

        ew.w1[e] = wbuf + off;   // [bn][512]
        {
            int total = REPR_DIM * bn;
            wt_convert<<<(total + 255) / 256, 256, 0, stream>>>(W1, wbuf + off, REPR_DIM, bn);
            off += (size_t)total;
        }
        ew.w2[e] = wbuf + off;   // [bn][bn]
        {
            int total = bn * bn;
            wt_convert<<<(total + 255) / 256, 256, 0, stream>>>(W2, wbuf + off, bn, bn);
            off += (size_t)total;
        }
        ew.w3[e] = wbuf + off;   // [512][bn]
        {
            int total = bn * REPR_DIM;
            wt_convert<<<(total + 255) / 256, 256, 0, stream>>>(W3, wbuf + off, bn, REPR_DIM);
            off += (size_t)total;
        }
    }

    gating_kernel<<<BATCH, 256, 0, stream>>>(x, wg, gates);
    loss_kernel<<<1, 256, 0, stream>>>(gates, out + (size_t)in_sizes[0]);  // loss after y
    moe_expert_kernel<<<NROWS / MTILE, 256, 0, stream>>>(x, gates, out, ew);
}
